// model_rel_gru_regressive_57861799411872
// MI455X (gfx1250) — compile-verified, matrix operand order checked by reading, not run on hardware
//
#include <hip/hip_runtime.h>
#include <math.h>

// Problem constants (from reference): HD,N,M,H,B,T,F,A = 4,64,32,256,256,8,12,1024 ; D=H+M=288
#define HDH   4
#define NSLOT 64
#define MW    32
#define HDIM  256
#define BSC   256
#define TT    8
#define FF    12
#define AQ    1024
#define DDIM  288
#define RMAX  (HDH*AQ)

typedef __bf16 bf16_t;
typedef bf16_t bf16x16 __attribute__((ext_vector_type(16)));
typedef float  floatx8 __attribute__((ext_vector_type(8)));
typedef unsigned int uint4v __attribute__((ext_vector_type(4)));
typedef int int4x __attribute__((ext_vector_type(4)));

#if defined(__has_builtin)
#if __has_builtin(__builtin_amdgcn_global_load_async_to_lds_b128)
#define HAVE_ASYNC_LDS 1
#endif
#endif
#ifndef HAVE_ASYNC_LDS
#define HAVE_ASYNC_LDS 0
#endif

__device__ __forceinline__ void wait_async0() {
#if __has_builtin(__builtin_amdgcn_s_wait_asynccnt)
  __builtin_amdgcn_s_wait_asynccnt(0);
#else
  asm volatile("s_wait_asynccnt 0x0" ::: "memory");
#endif
}

__device__ __forceinline__ unsigned short f2bf(float f) {
  unsigned int u = __float_as_uint(f);
  unsigned int r = u + 0x7FFFu + ((u >> 16) & 1u);   // round-to-nearest-even
  return (unsigned short)(r >> 16);
}
__device__ __forceinline__ float sigmoidf_(float x) { return 1.0f / (1.0f + __expf(-x)); }
__device__ __forceinline__ float softplusf_(float x) { return (x > 20.f) ? x : log1pf(__expf(x)); }

// ---------------------------------------------------------------------------
// WMMA GEMM:  Y[M x Npad] = Xbf[M x K] @ Wbf[Npad x K]^T + bias  (bf16 in, f32 acc)
// Block = 4 waves, computes a 64(M) x 64(N) C tile.  Per K-step (32):
//   - 64x32 weight tile staged into LDS (async-to-LDS when available),
//     shared by all 4 waves (4x weight-traffic reduction),
//   - each wave: 1 A fragment (its 16 rows) + 4 B fragments from LDS
//     -> 4 v_wmma_f32_16x16x32_bf16 into 4 accumulators.
// Fragment layout per ISA 16-bit A layout: lane l holds row (l&15); dword j
// holds K = (j>=4?16:0) + 8*(l>>4) + 2*(j&3) (+1). B lane-symmetric with W
// stored row-major (N,K)  (B column n = W row n).
// ---------------------------------------------------------------------------
__global__ void __launch_bounds__(128) gemm_bf16_wmma(
    float* __restrict__ Y, const unsigned short* __restrict__ X,
    const unsigned short* __restrict__ W, const float* __restrict__ bias,
    int K, int ldY, int n_valid,
    long sY, long sX, long sW, long sB)
{
  __shared__ unsigned short smem[64 * 32];
  int b = blockIdx.z;
  Y += sY * b; X += sX * b; W += sW * b;
  if (bias) bias += sB * b;
  int tid  = threadIdx.x;
  int w    = tid >> 5;
  int lane = tid & 31;
  int r    = lane & 15;
  int half = lane >> 4;
  int m0 = blockIdx.x * 64 + w * 16;
  int n0 = blockIdx.y * 64;
  const unsigned short* xrow = X + (long)(m0 + r) * K;
  floatx8 acc[4];
#pragma unroll
  for (int nt = 0; nt < 4; ++nt) acc[nt] = (floatx8){};

  for (int k0 = 0; k0 < K; k0 += 32) {
    // ---- stage 64(N) x 32(K) weight tile into LDS ----
    for (int c = tid; c < 256; c += 128) {      // 256 chunks of 16B
      int n  = c >> 2;
      int ko = (c & 3) << 3;
      const unsigned short* g = W + (long)(n0 + n) * K + k0 + ko;
      unsigned short* l = smem + n * 32 + ko;
#if HAVE_ASYNC_LDS
      __builtin_amdgcn_global_load_async_to_lds_b128(
          (__attribute__((address_space(1))) int4x*)g,
          (__attribute__((address_space(3))) int4x*)l, 0, 0);
#else
      *(uint4v*)l = *(const uint4v*)g;
#endif
    }
#if HAVE_ASYNC_LDS
    wait_async0();
#endif
    __syncthreads();

    // ---- A fragment (this wave's 16 rows) ----
    union { bf16x16 v; unsigned int u[8]; } fa, fb;
    const unsigned short* xr_ = xrow + k0;
#pragma unroll
    for (int j = 0; j < 8; ++j) {
      int kk = ((j & 4) << 2) + (half << 3) + ((j & 3) << 1);
      fa.u[j] = *(const unsigned int*)(xr_ + kk);
    }
    // ---- 4 B fragments from LDS, 4 WMMAs ----
#pragma unroll
    for (int nt = 0; nt < 4; ++nt) {
      const unsigned short* lrow = smem + (nt * 16 + r) * 32;
#pragma unroll
      for (int j = 0; j < 8; ++j) {
        int kk = ((j & 4) << 2) + (half << 3) + ((j & 3) << 1);
        fb.u[j] = *(const unsigned int*)(lrow + kk);
      }
      acc[nt] = __builtin_amdgcn_wmma_f32_16x16x32_bf16(false, fa.v, false, fb.v,
                                                        (short)0, acc[nt], false, false);
    }
    __syncthreads();
  }

  int row0 = m0 + half * 8;
#pragma unroll
  for (int nt = 0; nt < 4; ++nt) {
    int col = n0 + nt * 16 + r;
    float bv = (bias != nullptr && col < n_valid) ? bias[col] : 0.0f;
#pragma unroll
    for (int i = 0; i < 8; ++i)
      Y[(long)(row0 + i) * ldY + col] = acc[nt][i] + bv;
  }
}

// ---------------------------------------------------------------------------
// Weight conversion f32 -> bf16 with row (N) and column (K) zero padding,
// and transposed variant (W stored (K,N) -> (Npad,K)).
// ---------------------------------------------------------------------------
__global__ void conv_pad_kernel(unsigned short* dst, const float* src,
                                int rows, int rowspad, int cols, int colspad) {
  long i = (long)blockIdx.x * blockDim.x + threadIdx.x;
  long total = (long)rowspad * colspad;
  if (i >= total) return;
  int  c = (int)(i % colspad);
  long r = i / colspad;
  dst[i] = (r < rows && c < cols) ? f2bf(src[r * cols + c]) : (unsigned short)0;
}
__global__ void conv_padT_kernel(unsigned short* dst, const float* src, int K, int nvalid, int npad) {
  long i = (long)blockIdx.x * blockDim.x + threadIdx.x;
  long total = (long)npad * K;
  if (i >= total) return;
  int k = (int)(i % K);
  int n = (int)(i / K);
  dst[i] = (n < nvalid) ? f2bf(src[(long)k * nvalid + n]) : (unsigned short)0;
}

__global__ void zero_kernel(float* p, long n) {
  long i = (long)blockIdx.x * blockDim.x + threadIdx.x;
  if (i < n) p[i] = 0.f;
}

// dst[i] = src[i % period]; optional f32 dest, bf16 dest, and a second bf16 dest
// laid out with row length ld2 (rows of srccols columns, column offset off2).
__global__ void bcast_kernel(float* dstf, unsigned short* dstbf, const float* src,
                             long total, long period,
                             unsigned short* dstbf2, int srccols, int ld2, int off2) {
  long i = (long)blockIdx.x * blockDim.x + threadIdx.x;
  if (i >= total) return;
  float v = src[i % period];
  if (dstf) dstf[i] = v;
  unsigned short bv = f2bf(v);
  if (dstbf) dstbf[i] = bv;
  if (dstbf2) {
    long rr = i / srccols;
    int  c  = (int)(i % srccols);
    dstbf2[rr * ld2 + off2 + c] = bv;
  }
}

// broadcast a length-ncols f32 vector into bf16 columns [off, off+ncols) of R rows
__global__ void bcast_cols_kernel(unsigned short* dst, int ld, int off, const float* src, int ncols, int R) {
  int i = blockIdx.x * blockDim.x + threadIdx.x;
  if (i >= R * ncols) return;
  int r = i / ncols, c = i % ncols;
  dst[(long)r * ld + off + c] = f2bf(src[c]);
}

// ---------------------------------------------------------------------------
// tiny MLP: relu(x@w1+b1)@w2+b2  (2->8->16), writes bf16 cols [0,16) of dst,
// zero-fills cols [z0,z1) (the K padding of the concat buffer).
// ---------------------------------------------------------------------------
__global__ void mlp_kernel(unsigned short* dst, int ld, int z0, int z1,
                           const float* x, int rstride,
                           const float* w1, const float* b1,
                           const float* w2, const float* b2, int R) {
  int r = blockIdx.x * blockDim.x + threadIdx.x;
  if (r >= R) return;
  float x0 = x[(long)r * rstride];
  float x1 = x[(long)r * rstride + 1];
  float hb[8];
#pragma unroll
  for (int j = 0; j < 8; ++j) hb[j] = fmaxf(x0 * w1[j] + x1 * w1[8 + j] + b1[j], 0.f);
  unsigned short* out = dst + (long)r * ld;
#pragma unroll
  for (int j = 0; j < 16; ++j) {
    float o = b2[j];
#pragma unroll
    for (int i = 0; i < 8; ++i) o += hb[i] * w2[i * 16 + j];
    out[j] = f2bf(o);
  }
  for (int j = z0; j < z1; ++j) out[j] = 0;
}

// ---------------------------------------------------------------------------
// GRU combine: h' = (1-z)*tanh(inn + r*hn) + z*h ; writes f32 h, bf16 h, and
// optionally a second bf16 copy (into the concat buffer for the decoder GRU).
// ---------------------------------------------------------------------------
__global__ void gru_kernel(float* h, unsigned short* hbf, const float* gi, const float* gh,
                           int G, int ld3, long R, unsigned short* hbf2, int ld2, int off2) {
  long i = (long)blockIdx.x * blockDim.x + threadIdx.x;
  long total = R * G;
  if (i >= total) return;
  long r = i / G;
  int  g = (int)(i % G);
  const float* gir = gi + r * ld3;
  const float* ghr = gh + r * ld3;
  float ir = gir[g], iz = gir[G + g], inn = gir[2 * G + g];
  float hr = ghr[g], hz = ghr[G + g], hn  = ghr[2 * G + g];
  float rg = sigmoidf_(ir + hr);
  float z  = sigmoidf_(iz + hz);
  float nn = tanhf(inn + rg * hn);
  float hp = (1.f - z) * nn + z * h[i];
  h[i] = hp;
  unsigned short bv = f2bf(hp);
  hbf[i] = bv;
  if (hbf2) hbf2[r * ld2 + off2 + g] = bv;
}

// ---------------------------------------------------------------------------
// Cosine addressing + read:  w = softmax(beta * (k.mem_n)/(|k||mem_n|)),
// reads[m] = sum_n w[n] mem[n][m].  Block = 64 threads (one per slot).
// ---------------------------------------------------------------------------
__global__ void __launch_bounds__(64) read_kernel(
    const float* __restrict__ o, int ldo,
    const float* __restrict__ mem, long memStride,
    const int* __restrict__ scene, int A,
    unsigned short* dst1, int ld1, int off1,
    unsigned short* dst2, int ld2, int off2)
{
  int a = blockIdx.x, k = blockIdx.y;
  long rrow = (long)k * A + a;
  const float* orow = o + rrow * ldo;
  const float* memS = mem + (long)k * memStride + (long)scene[a] * (NSLOT * MW);
  __shared__ float skey[MW];
  __shared__ float sw[NSLOT];
  int tid = threadIdx.x;
  if (tid < MW) skey[tid] = orow[tid];
  __syncthreads();
  float beta = softplusf_(orow[MW]);
  float num = 0.f, mm2 = 0.f, kk = 0.f;
  const float* mrow = memS + tid * MW;
#pragma unroll
  for (int m = 0; m < MW; ++m) {
    float kv = skey[m], mv = mrow[m];
    num += kv * mv; mm2 += mv * mv; kk += kv * kv;
  }
  float den = fmaxf(sqrtf(kk) * sqrtf(mm2), 1e-8f);
  sw[tid] = beta * num / den;
  __syncthreads();
  float mx = -3.4e38f;
  for (int j = 0; j < NSLOT; ++j) mx = fmaxf(mx, sw[j]);
  float ssum = 0.f;
  for (int j = 0; j < NSLOT; ++j) ssum += __expf(sw[j] - mx);
  float wv = __expf(sw[tid] - mx) / ssum;
  __syncthreads();
  sw[tid] = wv;
  __syncthreads();
  if (tid < MW) {
    float rsum = 0.f;
    for (int j = 0; j < NSLOT; ++j) rsum += sw[j] * memS[j * MW + tid];
    unsigned short bv = f2bf(rsum);
    dst1[rrow * ld1 + off1 + tid] = bv;
    if (dst2) dst2[rrow * ld2 + off2 + tid] = bv;
  }
}

// ---------------------------------------------------------------------------
// Write addressing + segment-sum accumulation (atomics over scenes).
// ---------------------------------------------------------------------------
__global__ void __launch_bounds__(64) write_kernel(
    const float* __restrict__ o, int ldo,
    const float* __restrict__ mem, long memStride,
    const int* __restrict__ scene, int A,
    float* lacc, float* aacc, long accStride)
{
  int a = blockIdx.x, k = blockIdx.y;
  long rrow = (long)k * A + a;
  const float* orow = o + rrow * ldo;
  const float* memS = mem + (long)k * memStride + (long)scene[a] * (NSLOT * MW);
  __shared__ float skey[MW];
  __shared__ float se[MW];
  __shared__ float sa[MW];
  __shared__ float sw[NSLOT];
  int tid = threadIdx.x;
  if (tid < MW) {
    skey[tid] = orow[tid];
    se[tid]   = sigmoidf_(orow[MW + 1 + tid]);
    sa[tid]   = orow[2 * MW + 1 + tid];
  }
  __syncthreads();
  float beta = softplusf_(orow[MW]);
  float num = 0.f, mm2 = 0.f, kk = 0.f;
  const float* mrow = memS + tid * MW;
#pragma unroll
  for (int m = 0; m < MW; ++m) {
    float kv = skey[m], mv = mrow[m];
    num += kv * mv; mm2 += mv * mv; kk += kv * kv;
  }
  float den = fmaxf(sqrtf(kk) * sqrtf(mm2), 1e-8f);
  sw[tid] = beta * num / den;
  __syncthreads();
  float mx = -3.4e38f;
  for (int j = 0; j < NSLOT; ++j) mx = fmaxf(mx, sw[j]);
  float ssum = 0.f;
  for (int j = 0; j < NSLOT; ++j) ssum += __expf(sw[j] - mx);
  float ww = __expf(sw[tid] - mx) / ssum;
  int s = scene[a];
  float* lp = lacc + (long)k * accStride + ((long)s * NSLOT + tid) * MW;
  float* ap = aacc + (long)k * accStride + ((long)s * NSLOT + tid) * MW;
  for (int m = 0; m < MW; ++m) {
    atomicAdd(&lp[m], log1pf(-ww * se[m]));
    atomicAdd(&ap[m], ww * sa[m]);
  }
}

// mem = mem*exp(lacc) + aacc ; re-zeroes the accumulators for the next step.
__global__ void mem_update_kernel(float* mem, float* lacc, float* aacc, long n) {
  long i = (long)blockIdx.x * blockDim.x + threadIdx.x;
  if (i >= n) return;
  mem[i] = mem[i] * __expf(lacc[i]) + aacc[i];
  lacc[i] = 0.f;
  aacc[i] = 0.f;
}

// out = hfut @ fco_w + fco_b ; pres update ; scatter into output (A, HD, F, 2)
__global__ void out_pres_kernel(float* pred, float* pres, float* outbuf, const float* hfut,
                                const float* fcow, const float* fcob, const float* past, int t) {
  int rr = blockIdx.x * blockDim.x + threadIdx.x;
  if (rr >= RMAX) return;
  const float* hv = hfut + (long)rr * DDIM;
  float o0 = fcob[0], o1 = fcob[1];
  for (int c = 0; c < DDIM; ++c) {
    float x = hv[c];
    o0 += x * fcow[2 * c];
    o1 += x * fcow[2 * c + 1];
  }
  int a = rr % AQ, k = rr / AQ;
  float p0, p1;
  if (t == 0) {
    p0 = past[a * TT * 2 + (TT - 1) * 2 + 0] + o0;
    p1 = past[a * TT * 2 + (TT - 1) * 2 + 1] + o1;
  } else {
    p0 = pres[rr * 2] + o0;
    p1 = pres[rr * 2 + 1] + o1;
  }
  pres[rr * 2] = p0; pres[rr * 2 + 1] = p1;
  outbuf[rr * 2] = o0; outbuf[rr * 2 + 1] = o1;
  long oidx = (((long)a * HDH + k) * FF + t) * 2;
  pred[oidx] = p0; pred[oidx + 1] = p1;
}

// ===========================================================================
static inline unsigned gdim(long n) { return (unsigned)((n + 255) / 256); }

extern "C" void kernel_launch(void* const* d_in, const int* in_sizes, int n_in,
                              void* d_out, int out_size, void* d_ws, size_t ws_size,
                              hipStream_t stream) {
  (void)in_sizes; (void)n_in; (void)out_size;
  const float* past     = (const float*)d_in[0];
  const float* past_rel = (const float*)d_in[1];
  const float* ep_w1 = (const float*)d_in[2];
  const float* ep_b1 = (const float*)d_in[3];
  const float* ep_w2 = (const float*)d_in[4];
  const float* ep_b2 = (const float*)d_in[5];
  const float* er_w1 = (const float*)d_in[6];
  const float* er_b1 = (const float*)d_in[7];
  const float* er_w2 = (const float*)d_in[8];
  const float* er_b2 = (const float*)d_in[9];
  const float* enc_wih = (const float*)d_in[10];
  const float* enc_whh = (const float*)d_in[11];
  const float* enc_bih = (const float*)d_in[12];
  const float* enc_bhh = (const float*)d_in[13];
  const float* encr_wih = (const float*)d_in[14];
  const float* encr_whh = (const float*)d_in[15];
  const float* encr_bih = (const float*)d_in[16];
  const float* encr_bhh = (const float*)d_in[17];
  const float* dec_wih = (const float*)d_in[18];
  const float* dec_whh = (const float*)d_in[19];
  const float* dec_bih = (const float*)d_in[20];
  const float* dec_bhh = (const float*)d_in[21];
  const float* fco_w = (const float*)d_in[22];
  const float* fco_b = (const float*)d_in[23];
  const float* fr_w  = (const float*)d_in[24];
  const float* fr_b  = (const float*)d_in[25];
  const float* fw_w  = (const float*)d_in[26];
  const float* fw_b  = (const float*)d_in[27];
  const float* frp_w = (const float*)d_in[28];
  const float* frp_b = (const float*)d_in[29];
  const float* fwp_w = (const float*)d_in[30];
  const float* fwp_b = (const float*)d_in[31];
  const float* h_bias     = (const float*)d_in[32];
  const float* h_bias_rel = (const float*)d_in[33];
  const float* h_bias_fut = (const float*)d_in[34];
  const float* init_r     = (const float*)d_in[35];
  const float* mem_bias   = (const float*)d_in[36];
  const int*   scene      = (const int*)d_in[37];
  float* pred = (float*)d_out;

  // ---- workspace carve-out -------------------------------------------------
  char* wp = (char*)d_ws;
  auto alloc = [&](size_t bytes) -> void* {
    void* p = (void*)wp;
    wp += (bytes + 255) & ~(size_t)255;
    return p;
  };
  unsigned short* encr_wih_bf = (unsigned short*)alloc((size_t)768 * 32 * 2);
  unsigned short* encr_whh_bf = (unsigned short*)alloc((size_t)768 * 256 * 2);
  unsigned short* enc_wih_bf  = (unsigned short*)alloc((size_t)768 * 64 * 2);
  unsigned short* enc_whh_bf  = (unsigned short*)alloc((size_t)768 * 256 * 2);
  unsigned short* dec_wih_bf  = (unsigned short*)alloc((size_t)896 * 288 * 2);
  unsigned short* dec_whh_bf  = (unsigned short*)alloc((size_t)896 * 288 * 2);
  unsigned short* frp_bf      = (unsigned short*)alloc((size_t)64 * 256 * 2);
  unsigned short* fwp_bf      = (unsigned short*)alloc((size_t)128 * 256 * 2);
  unsigned short* fr_bf       = (unsigned short*)alloc((size_t)HDH * 64 * 256 * 2);
  unsigned short* fw_bf       = (unsigned short*)alloc((size_t)HDH * 128 * 256 * 2);

  float* gi    = (float*)alloc((size_t)RMAX * 896 * 4);
  float* gh    = (float*)alloc((size_t)RMAX * 896 * 4);
  float* obuf  = (float*)alloc((size_t)RMAX * 64 * 4);
  float* owbuf = (float*)alloc((size_t)RMAX * 128 * 4);

  unsigned short* xr_bf   = (unsigned short*)alloc((size_t)RMAX * 32 * 2);
  unsigned short* x_bf    = (unsigned short*)alloc((size_t)RMAX * 64 * 2);
  unsigned short* xfut_bf = (unsigned short*)alloc((size_t)RMAX * DDIM * 2);

  float* hbuf   = (float*)alloc((size_t)AQ * HDIM * 4);
  unsigned short* hbf = (unsigned short*)alloc((size_t)AQ * HDIM * 2);
  float* hrel   = (float*)alloc((size_t)AQ * HDIM * 4);
  unsigned short* hrelbf = (unsigned short*)alloc((size_t)AQ * HDIM * 2);

  float* hHf = (float*)alloc((size_t)RMAX * HDIM * 4);
  unsigned short* hHbf = (unsigned short*)alloc((size_t)RMAX * HDIM * 2);
  float* hrelHf = (float*)alloc((size_t)RMAX * HDIM * 4);
  unsigned short* hrelHbf = (unsigned short*)alloc((size_t)RMAX * HDIM * 2);
  float* hfutf = (float*)alloc((size_t)RMAX * DDIM * 4);
  unsigned short* hfutbf = (unsigned short*)alloc((size_t)RMAX * DDIM * 2);

  const long BNM = (long)BSC * NSLOT * MW;
  float* memb = (float*)alloc((size_t)BNM * 4);
  float* memH = (float*)alloc((size_t)HDH * BNM * 4);
  float* lacc = (float*)alloc((size_t)HDH * BNM * 4);
  float* aacc = (float*)alloc((size_t)HDH * BNM * 4);
  float* pres = (float*)alloc((size_t)RMAX * 2 * 4);
  float* outb = (float*)alloc((size_t)RMAX * 2 * 4);

  if ((size_t)(wp - (char*)d_ws) > ws_size) return;  // workspace too small

  // ---- launch helpers ------------------------------------------------------
  auto gemm = [&](float* Y, int ldY, const unsigned short* X, const unsigned short* W,
                  const float* bias, int nvalid, int Mrows, int Npad, int K,
                  int batch, long sY, long sX, long sW, long sB) {
    dim3 grid(Mrows / 64, Npad / 64, batch);
    gemm_bf16_wmma<<<grid, dim3(128, 1, 1), 0, stream>>>(Y, X, W, bias, K, ldY, nvalid, sY, sX, sW, sB);
  };
  auto conv = [&](unsigned short* dst, const float* src, int rows, int rowspad, int cols, int colspad) {
    long total = (long)rowspad * colspad;
    conv_pad_kernel<<<gdim(total), 256, 0, stream>>>(dst, src, rows, rowspad, cols, colspad);
  };
  auto convT = [&](unsigned short* dst, const float* src, int K, int nv, int np) {
    long total = (long)np * K;
    conv_padT_kernel<<<gdim(total), 256, 0, stream>>>(dst, src, K, nv, np);
  };
  auto bcast = [&](float* dstf, unsigned short* dstbf, const float* src, long total, long period,
                   unsigned short* dstbf2, int srccols, int ld2, int off2) {
    bcast_kernel<<<gdim(total), 256, 0, stream>>>(dstf, dstbf, src, total, period, dstbf2, srccols, ld2, off2);
  };
  auto gru = [&](float* h, unsigned short* hb, int G, int ld3, long R,
                 unsigned short* hb2, int ld2, int off2) {
    gru_kernel<<<gdim(R * G), 256, 0, stream>>>(h, hb, gi, gh, G, ld3, R, hb2, ld2, off2);
  };
  auto mlp = [&](unsigned short* dst, int ld, int z0, int z1, const float* x, int rstride,
                 const float* w1, const float* b1, const float* w2, const float* b2, int R) {
    mlp_kernel<<<gdim(R), 256, 0, stream>>>(dst, ld, z0, z1, x, rstride, w1, b1, w2, b2, R);
  };

  // ---- weight conversion (bf16, padded) -----------------------------------
  conv(encr_wih_bf, encr_wih, 768, 768, 16, 32);
  conv(encr_whh_bf, encr_whh, 768, 768, 256, 256);
  conv(enc_wih_bf,  enc_wih,  768, 768, 48, 64);
  conv(enc_whh_bf,  enc_whh,  768, 768, 256, 256);
  conv(dec_wih_bf,  dec_wih,  864, 896, 288, 288);
  conv(dec_whh_bf,  dec_whh,  864, 896, 288, 288);
  convT(frp_bf, frp_w, 256, 33, 64);
  convT(fwp_bf, fwp_w, 256, 97, 128);
  for (int k = 0; k < HDH; ++k) {
    convT(fr_bf + (size_t)k * 64 * 256,  fr_w + (size_t)k * 256 * 33, 256, 33, 64);
    convT(fw_bf + (size_t)k * 128 * 256, fw_w + (size_t)k * 256 * 97, 256, 97, 128);
  }

  // ---- state init ----------------------------------------------------------
  zero_kernel<<<gdim(HDH * BNM), 256, 0, stream>>>(lacc, HDH * BNM);
  zero_kernel<<<gdim(HDH * BNM), 256, 0, stream>>>(aacc, HDH * BNM);
  bcast(hbuf, hbf, h_bias, (long)AQ * HDIM, HDIM, nullptr, 0, 0, 0);
  bcast(hrel, hrelbf, h_bias_rel, (long)AQ * HDIM, HDIM, nullptr, 0, 0, 0);
  bcast(memb, nullptr, mem_bias, BNM, NSLOT * MW, nullptr, 0, 0, 0);
  bcast_cols_kernel<<<gdim((long)AQ * MW), 256, 0, stream>>>(x_bf, 64, 16, init_r, MW, AQ);

  // ---- encoder: 7 scan steps + final GRU update ---------------------------
  for (int t = 0; t < TT; ++t) {
    // h_rel GRU
    mlp(xr_bf, 32, 16, 32, past_rel + t * 2, TT * 2, er_w1, er_b1, er_w2, er_b2, AQ);
    gemm(gi, 768, xr_bf, encr_wih_bf, encr_bih, 768, AQ, 768, 32, 1, 0, 0, 0, 0);
    gemm(gh, 768, hrelbf, encr_whh_bf, encr_bhh, 768, AQ, 768, 256, 1, 0, 0, 0, 0);
    gru(hrel, hrelbf, HDIM, 768, AQ, nullptr, 0, 0);
    // h GRU (concat([mlp(xp), reads]) already staged in x_bf)
    mlp(x_bf, 64, 48, 64, past + t * 2, TT * 2, ep_w1, ep_b1, ep_w2, ep_b2, AQ);
    gemm(gi, 768, x_bf, enc_wih_bf, enc_bih, 768, AQ, 768, 64, 1, 0, 0, 0, 0);
    gemm(gh, 768, hbf, enc_whh_bf, enc_bhh, 768, AQ, 768, 256, 1, 0, 0, 0, 0);
    gru(hbuf, hbf, HDIM, 768, AQ, nullptr, 0, 0);
    if (t == TT - 1) break;  // final step: no memory read/write
    // read (stages next step's reads into x_bf cols 16..47)
    gemm(obuf, 64, hbf, frp_bf, frp_b, 33, AQ, 64, 256, 1, 0, 0, 0, 0);
    read_kernel<<<dim3(AQ, 1, 1), 64, 0, stream>>>(obuf, 64, memb, 0, scene, AQ,
                                                   x_bf, 64, 16, nullptr, 0, 0);
    // write
    gemm(owbuf, 128, hbf, fwp_bf, fwp_b, 97, AQ, 128, 256, 1, 0, 0, 0, 0);
    write_kernel<<<dim3(AQ, 1, 1), 64, 0, stream>>>(owbuf, 128, memb, 0, scene, AQ, lacc, aacc, 0);
    mem_update_kernel<<<gdim(BNM), 256, 0, stream>>>(memb, lacc, aacc, BNM);
  }

  // ---- broadcast to heads --------------------------------------------------
  bcast(hHf, hHbf, hbuf, (long)RMAX * HDIM, (long)AQ * HDIM, nullptr, 0, 0, 0);
  bcast(hrelHf, hrelHbf, hrel, (long)RMAX * HDIM, (long)AQ * HDIM, xfut_bf, HDIM, DDIM, 0);
  bcast(hfutf, hfutbf, h_bias_fut, (long)RMAX * DDIM, DDIM, nullptr, 0, 0, 0);
  bcast(memH, nullptr, memb, HDH * BNM, BNM, nullptr, 0, 0, 0);

  // ---- head step (used F times) -------------------------------------------
  auto head_step = [&](int t) {
    gemm(obuf, 64, hHbf, fr_bf, fr_b, 33, AQ, 64, 256,
         HDH, (long)AQ * 64, (long)AQ * 256, (long)64 * 256, 33);
    read_kernel<<<dim3(AQ, HDH, 1), 64, 0, stream>>>(obuf, 64, memH, BNM, scene, AQ,
                                                     xfut_bf, DDIM, HDIM, x_bf, 64, 16);
    gemm(gi, 896, xfut_bf, dec_wih_bf, dec_bih, 864, RMAX, 896, DDIM, 1, 0, 0, 0, 0);
    gemm(gh, 896, hfutbf, dec_whh_bf, dec_bhh, 864, RMAX, 896, DDIM, 1, 0, 0, 0, 0);
    gru(hfutf, hfutbf, DDIM, 896, RMAX, nullptr, 0, 0);
    out_pres_kernel<<<gdim(RMAX), 256, 0, stream>>>(pred, pres, outb, hfutf, fco_w, fco_b, past, t);
    gemm(owbuf, 128, hHbf, fw_bf, fw_b, 97, AQ, 128, 256,
         HDH, (long)AQ * 128, (long)AQ * 256, (long)128 * 256, 97);
    write_kernel<<<dim3(AQ, HDH, 1), 64, 0, stream>>>(owbuf, 128, memH, BNM, scene, AQ, lacc, aacc, BNM);
    mem_update_kernel<<<gdim(HDH * BNM), 256, 0, stream>>>(memH, lacc, aacc, HDH * BNM);
  };

  head_step(0);

  // ---- decoder scan: F-1 steps --------------------------------------------
  for (int t = 1; t < FF; ++t) {
    // h_relH GRU from pres_rel (= previous out), writes bf16 into xfut cols 0..255
    mlp(xr_bf, 32, 16, 32, outb, 2, er_w1, er_b1, er_w2, er_b2, RMAX);
    gemm(gi, 768, xr_bf, encr_wih_bf, encr_bih, 768, RMAX, 768, 32, 1, 0, 0, 0, 0);
    gemm(gh, 768, hrelHbf, encr_whh_bf, encr_bhh, 768, RMAX, 768, 256, 1, 0, 0, 0, 0);
    gru(hrelHf, hrelHbf, HDIM, 768, RMAX, xfut_bf, DDIM, 0);
    // hH GRU from concat([mlp(pres), reading]) (reading staged by previous head_step)
    mlp(x_bf, 64, 48, 64, pres, 2, ep_w1, ep_b1, ep_w2, ep_b2, RMAX);
    gemm(gi, 768, x_bf, enc_wih_bf, enc_bih, 768, RMAX, 768, 64, 1, 0, 0, 0, 0);
    gemm(gh, 768, hHbf, enc_whh_bf, enc_bhh, 768, RMAX, 768, 256, 1, 0, 0, 0, 0);
    gru(hHf, hHbf, HDIM, 768, RMAX, nullptr, 0, 0);
    head_step(t);
  }
}
